// TIBlock_85753317032445
// MI455X (gfx1250) — matrix operand order repaired and checked by reading, not compiled, not run on hardware
//
#include <hip/hip_runtime.h>
#include <stdint.h>

// ---------------------------------------------------------------------------
// MI455X (gfx1250) implementation, bf16-resident pipeline, v3.
//  * All matmuls -> V_WMMA_F32_16X16X32_BF16 (wave32, f32 accumulate)
//  * GEMM: single-wave workgroups (barriers are S_NOP; same-wave LDS ordering
//    is guaranteed), 64x64 C tile per wave = 16 WMMA / K-step,
//    double-buffered LDS fed by global_load_async_to_lds_b128 (ASYNCcnt
//    in-order completion provides the pipeline fence).
//  * f32 kept where accuracy matters: accumulators, GLU pre-activation,
//    softmax logits, attention-output accumulation, pooling, head.
// Workspace requirement: ~259 MB.
// ---------------------------------------------------------------------------

#define USE_ASYNC_LDS 1

typedef unsigned short u16;
typedef unsigned int   u32;
typedef __attribute__((ext_vector_type(16))) __bf16 v16bf;
typedef __attribute__((ext_vector_type(8)))  float  v8f;

__device__ __forceinline__ u16 f2bf(float f) {
    u32 u = __builtin_bit_cast(u32, f);
    return (u16)((u + 0x7FFFu + ((u >> 16) & 1u)) >> 16);   // RNE
}
__device__ __forceinline__ float sigmoidf_(float x) {
    return 1.0f / (1.0f + __expf(-x));
}

// 16B global(bf16) -> LDS, contiguous
__device__ __forceinline__ void stage16(const u16* __restrict__ src, u16* dst) {
#if USE_ASYNC_LDS
    asm volatile("global_load_async_to_lds_b128 %0, %1, off"
                 :: "v"((u32)(uintptr_t)dst), "v"((unsigned long long)(uintptr_t)src)
                 : "memory");
#else
    *(uint4*)dst = *(const uint4*)src;
#endif
}
template <int Ncnt>
__device__ __forceinline__ void async_wait() {
#if USE_ASYNC_LDS
    asm volatile("s_wait_asynccnt %0" :: "i"(Ncnt) : "memory");
#endif
}

// Stage one 64x32 bf16 tile into LDS (row stride 40 halves = 80B, keeps every
// 16B chunk 16B-aligned).  TR=false: source rows contiguous in k -> async
// b128 copies.  TR=true: source is [k][row] -> b128 load + b16 scatter
// (same-wave DS ordering, no fence needed).
template <bool TR>
__device__ __forceinline__ void stage_tile(const u16* __restrict__ P, int ld,
                                           int org, int kk, u16* __restrict__ lbuf,
                                           int L)
{
    if (!TR) {
#pragma unroll
        for (int j = 0; j < 8; ++j) {
            const int q = L + j * 32;            // 0..255 chunks
            const int row = q >> 2, kq = (q & 3) * 8;
            stage16(P + (size_t)(org + row) * ld + kk + kq, &lbuf[row * 40 + kq]);
        }
    } else {
#pragma unroll
        for (int j = 0; j < 8; ++j) {
            const int q = L + j * 32;
            const int kc = q >> 3, rq = (q & 7) * 8;
            uint4 d = *(const uint4*)(P + (size_t)(kk + kc) * ld + org + rq);
            const u16* s16 = (const u16*)&d;
#pragma unroll
            for (int e = 0; e < 8; ++e) lbuf[(rq + e) * 40 + kc] = s16[e];
        }
    }
}

// ---------------------------------------------------------------------------
// Batched GEMM, bf16 in / (f32 or bf16) out.  M%64==0, N%64==0, K%32==0.
// TA: A[m,k]=A[k*lda+m]  else A[m*lda+k]
// TB: B[k,n]=B[n*ldb+k]  else B[k*ldb+n]
// One wave (32 threads) per block; 64x64 C tile; 16 WMMA per K-step.
// ---------------------------------------------------------------------------
template <bool TA, bool TB, bool ACC, bool OBF>
__global__ __launch_bounds__(32) void gemm_k(
    const u16* __restrict__ A, int lda, long long sA,
    const u16* __restrict__ B, int ldb, long long sB,
    void* __restrict__ Cv, int ldc, long long sC,
    int K, const float* __restrict__ bias_m)
{
    constexpr int APT = (TA ? 0 : 8) + (TB ? 8 : 0);   // async ops per tile
    const int L   = threadIdx.x & 31;
    const int hi  = (L >> 4) & 1, l16 = L & 15;
    const int m0  = blockIdx.y * 64, n0 = blockIdx.x * 64;

    A += (size_t)blockIdx.z * (size_t)sA;
    B += (size_t)blockIdx.z * (size_t)sB;

    __shared__ __align__(16) u16 lA[2][64 * 40];
    __shared__ __align__(16) u16 lB[2][64 * 40];

    v8f zero;
#pragma unroll
    for (int r = 0; r < 8; ++r) zero[r] = 0.0f;
    v8f acc[4][4];
#pragma unroll
    for (int mt = 0; mt < 4; ++mt)
#pragma unroll
        for (int nt = 0; nt < 4; ++nt) acc[mt][nt] = zero;

    // prologue: stage K-slab 0 into buffer 0
    stage_tile<TA>(A, lda, m0, 0, lA[0], L);
    stage_tile<!TB>(B, ldb, n0, 0, lB[0], L);

    int cur = 0;
    for (int kk = 0; kk < K; kk += 32) {
        if (kk + 32 < K) {   // stream next slab into the other buffer
            stage_tile<TA>(A, lda, m0, kk + 32, lA[cur ^ 1], L);
            stage_tile<!TB>(B, ldb, n0, kk + 32, lB[cur ^ 1], L);
            if constexpr (APT > 0) async_wait<APT>();   // current slab landed
        } else {
            if constexpr (APT > 0) async_wait<0>();
        }

        union Frag { u32 u[8]; v16bf v; };
        // B fragments (this code's k order for B 32x16, same as A: lanes0-15 K=2v,2v+1; lanes16-31 K+=16)
        Frag fb[4];
#pragma unroll
        for (int nt = 0; nt < 4; ++nt)
#pragma unroll
            for (int v = 0; v < 8; ++v)
                fb[nt].u[v] = *(const u32*)&lB[cur][(nt * 16 + l16) * 40 + 2 * v + hi * 16];

#pragma unroll
        for (int mt = 0; mt < 4; ++mt) {
            // A fragment (16-bit A 16x32: K=2v,2v+1 (+16*hi), same order as B)
            Frag fa;
#pragma unroll
            for (int v = 0; v < 8; ++v) {
                const int kp = 2 * v + hi * 16;
                fa.u[v] = *(const u32*)&lA[cur][(mt * 16 + l16) * 40 + kp];
            }
#pragma unroll
            for (int nt = 0; nt < 4; ++nt)
                acc[mt][nt] = __builtin_amdgcn_wmma_f32_16x16x32_bf16(
                    false, fa.v, false, fb[nt].v, (short)0, acc[mt][nt], false, false);
        }
        cur ^= 1;
    }

    // epilogue (C/D layout: lanes0-15 M=r, lanes16-31 M=r+8; N = lane%16)
    if (OBF) {
        u16* C = (u16*)Cv + (size_t)blockIdx.z * (size_t)sC;
#pragma unroll
        for (int mt = 0; mt < 4; ++mt)
#pragma unroll
            for (int nt = 0; nt < 4; ++nt)
#pragma unroll
                for (int r = 0; r < 8; ++r) {
                    const int gm = m0 + mt * 16 + r + hi * 8;
                    const int gn = n0 + nt * 16 + l16;
                    float v = acc[mt][nt][r];
                    if (bias_m) v += bias_m[gm];
                    C[(size_t)gm * ldc + gn] = f2bf(v);
                }
    } else {
        float* C = (float*)Cv + (size_t)blockIdx.z * (size_t)sC;
#pragma unroll
        for (int mt = 0; mt < 4; ++mt)
#pragma unroll
            for (int nt = 0; nt < 4; ++nt)
#pragma unroll
                for (int r = 0; r < 8; ++r) {
                    const int gm = m0 + mt * 16 + r + hi * 8;
                    const int gn = n0 + nt * 16 + l16;
                    const size_t off = (size_t)gm * ldc + gn;
                    float v = acc[mt][nt][r];
                    if (bias_m) v += bias_m[gm];
                    if (ACC) v += C[off];
                    C[off] = v;
                }
    }
}

// ---------------------------------------------------------------------------
// Support kernels (unchanged from v2)
// ---------------------------------------------------------------------------
__global__ void cvt_k(const float* __restrict__ in, u16* __restrict__ out, long long q)
{
    long long i = (long long)blockIdx.x * blockDim.x + threadIdx.x;
    if (i >= q) return;
    float4 v = ((const float4*)in)[i];
    uint2 r;
    r.x = (u32)f2bf(v.x) | ((u32)f2bf(v.y) << 16);
    r.y = (u32)f2bf(v.z) | ((u32)f2bf(v.w) << 16);
    ((uint2*)out)[i] = r;
}

__global__ void glu_last_k(const float* __restrict__ y, const float* __restrict__ b,
                           u16* __restrict__ out, int BN, int F)
{
    int i = blockIdx.x * blockDim.x + threadIdx.x;
    if (i >= BN * F) return;
    int row = i / F, c = i % F;
    float o = y[(size_t)row * 2 * F + c] + b[c];
    float g = y[(size_t)row * 2 * F + F + c] + b[F + c];
    out[i] = f2bf(o * sigmoidf_(g));
}

__global__ void glu_ch_k(const float* __restrict__ y, const float* __restrict__ b,
                         u16* __restrict__ out, int Bc, int Co2, int N)
{
    const int Co = Co2 >> 1;
    long long i = (long long)blockIdx.x * blockDim.x + threadIdx.x;
    if (i >= (long long)Bc * Co * N) return;
    int n = (int)(i % N);
    long long t = i / N;
    int c = (int)(t % Co), bb = (int)(t / Co);
    float o = y[((size_t)bb * Co2 + c) * N + n] + b[c];
    float g = y[((size_t)bb * Co2 + Co + c) * N + n] + b[Co + c];
    out[i] = f2bf(o * sigmoidf_(g));
}

__global__ void im2col_k(const u16* __restrict__ in, u16* __restrict__ col,
                         int Bc, int Ci, int N, int chLast)
{
    long long i = (long long)blockIdx.x * blockDim.x + threadIdx.x;
    if (i >= (long long)Bc * Ci * 7 * N) return;
    int n = (int)(i % N);
    long long t1 = i / N;
    int r = (int)(t1 % (Ci * 7));
    int b = (int)(t1 / (Ci * 7));
    int ci = r / 7, t = r % 7;
    int p = n + t - 3;
    u16 v = 0;
    if (p >= 0 && p < N)
        v = chLast ? in[((size_t)b * N + p) * Ci + ci]
                   : in[((size_t)b * Ci + ci) * N + p];
    col[i] = v;
}

// row softmax (f32 logits) + post-softmax scale, bf16 packed in place
__global__ __launch_bounds__(256) void softmax_pack_k(float* __restrict__ S,
                                                      int N, float scale)
{
    float* p = S + (size_t)blockIdx.x * N;
    __shared__ float red[256];
    const int tid = threadIdx.x;
    float m = -3.4e38f;
    for (int n = tid; n < N; n += 256) m = fmaxf(m, p[n]);
    red[tid] = m; __syncthreads();
    for (int s = 128; s > 0; s >>= 1) {
        if (tid < s) red[tid] = fmaxf(red[tid], red[tid + s]);
        __syncthreads();
    }
    m = red[0]; __syncthreads();
    float lv[4];
    int cnt = 0;
    float sum = 0.0f;
    for (int n = tid; n < N; n += 256) { float e = __expf(p[n] - m); lv[cnt++] = e; sum += e; }
    red[tid] = sum; __syncthreads();
    for (int s = 128; s > 0; s >>= 1) {
        if (tid < s) red[tid] += red[tid + s];
        __syncthreads();
    }
    const float inv = scale / red[0];
    __syncthreads();
    u16* o = (u16*)p;
    cnt = 0;
    for (int n = tid; n < N; n += 256) o[n] = f2bf(lv[cnt++] * inv);
}

__global__ __launch_bounds__(128) void mean_pool_k(const float* __restrict__ F,
                                                   float* __restrict__ P, int N, int C)
{
    const int c = blockIdx.x, b = blockIdx.y, tid = threadIdx.x;
    __shared__ float red[128];
    float s = 0.0f;
    for (int n = tid; n < N; n += 128) s += F[((size_t)b * N + n) * C + c];
    red[tid] = s; __syncthreads();
    for (int st = 64; st > 0; st >>= 1) {
        if (tid < st) red[tid] += red[tid + st];
        __syncthreads();
    }
    if (tid == 0) P[(size_t)b * C + c] = red[0] / (float)N;
}

__global__ void head_k(const float* __restrict__ pooled,
                       const float* __restrict__ w1, const float* __restrict__ b1,
                       const float* __restrict__ w2, const float* __restrict__ b2,
                       const float* __restrict__ eps,
                       float* __restrict__ out, int Bc, int C)
{
    int i = blockIdx.x * blockDim.x + threadIdx.x;
    if (i >= Bc * C) return;
    int b = i / C, c = i % C;
    const float* pv = pooled + (size_t)b * C;
    float a1 = b1[c], a2 = b2[c];
    for (int k = 0; k < C; ++k) {
        float p = pv[k];
        a1 += p * w1[(size_t)c * C + k];
        a2 += p * w2[(size_t)c * C + k];
    }
    a1 = fmaxf(a1, 0.0f);
    a2 = fmaxf(a2, 0.0f);
    float sd = __expf(0.5f * a2);
    out[i] = eps[i] * sd + a1;
    out[Bc * C + i] = a1;
    out[2 * Bc * C + i] = a2;
}

// ---------------------------------------------------------------------------
// Host-side helpers
// ---------------------------------------------------------------------------
static inline void launch_gemm(hipStream_t st, bool ta, bool tb, bool accum, bool obf,
                               const u16* A, int lda, long long sA,
                               const u16* B, int ldb, long long sB,
                               void* C, int ldc, long long sC,
                               int M, int N, int K, int batch, const float* bias)
{
    dim3 g(N / 64, M / 64, batch), blk(32);
    if (!ta && !tb && !accum && obf)
        gemm_k<false, false, false, true ><<<g, blk, 0, st>>>(A, lda, sA, B, ldb, sB, C, ldc, sC, K, bias);
    else if (!ta && !tb && !accum)
        gemm_k<false, false, false, false><<<g, blk, 0, st>>>(A, lda, sA, B, ldb, sB, C, ldc, sC, K, bias);
    else if (!ta && tb && !accum && obf)
        gemm_k<false, true,  false, true ><<<g, blk, 0, st>>>(A, lda, sA, B, ldb, sB, C, ldc, sC, K, bias);
    else if (ta)
        gemm_k<true,  false, false, false><<<g, blk, 0, st>>>(A, lda, sA, B, ldb, sB, C, ldc, sC, K, bias);
    else if (!ta && tb && !accum)
        gemm_k<false, true,  false, false><<<g, blk, 0, st>>>(A, lda, sA, B, ldb, sB, C, ldc, sC, K, bias);
    else
        gemm_k<false, true,  true,  false><<<g, blk, 0, st>>>(A, lda, sA, B, ldb, sB, C, ldc, sC, K, bias);
}
static inline void cvt(hipStream_t st, const float* in, u16* out, long long n)
{
    long long q = n / 4;
    cvt_k<<<(unsigned)((q + 255) / 256), 256, 0, st>>>(in, out, q);
}

extern "C" void kernel_launch(void* const* d_in, const int* in_sizes, int n_in,
                              void* d_out, int out_size, void* d_ws, size_t ws_size,
                              hipStream_t stream)
{
    (void)in_sizes; (void)n_in; (void)out_size; (void)ws_size;
    const int B = 16, N = 1024, C3 = 192;
    const float NORM = 0.07216878364870322f;            // 1/sqrt(192)
    const size_t NN = (size_t)N * N;
    const size_t MBb = 1024u * 1024u;

    const float* x      = (const float*)d_in[0];
    const float* adj    = (const float*)d_in[1];
    const float* degree = (const float*)d_in[2];
    const float* eps    = (const float*)d_in[3];
    const float* gW[3]  = {(const float*)d_in[4], (const float*)d_in[6], (const float*)d_in[8]};
    const float* gB[3]  = {(const float*)d_in[5], (const float*)d_in[7], (const float*)d_in[9]};
    const float* cW[3]  = {(const float*)d_in[10], (const float*)d_in[12], (const float*)d_in[14]};
    const float* cB[3]  = {(const float*)d_in[11], (const float*)d_in[13], (const float*)d_in[15]};
    const float* pW[6]  = {(const float*)d_in[16], (const float*)d_in[18], (const float*)d_in[20],
                           (const float*)d_in[22], (const float*)d_in[24], (const float*)d_in[26]};
    const float* pB[6]  = {(const float*)d_in[17], (const float*)d_in[19], (const float*)d_in[21],
                           (const float*)d_in[23], (const float*)d_in[25], (const float*)d_in[27]};
    const float* l1_w = (const float*)d_in[28]; const float* l1_b = (const float*)d_in[29];
    const float* l2_w = (const float*)d_in[30]; const float* l2_b = (const float*)d_in[31];
    float* out = (float*)d_out;
    char*  wsb = (char*)d_ws;

    // ----- workspace layout (byte offsets), ~259 MB -----------------------
    u16* ADJ_BF  = (u16*)(wsb + 0 * MBb);      // 32MB  } dead after T1/ADJN:
    u16* DEG_BF  = (u16*)(wsb + 32 * MBb);     // 32MB  }  reused as SSR (f32)
    u16* T1_BF   = (u16*)(wsb + 64 * MBb);     // 32MB  } dead after ADJN/GCN:
    u16* ADJN_BF = (u16*)(wsb + 96 * MBb);     // 32MB  }  reused as SDR (f32)
    u16* X_BF    = (u16*)(wsb + 128 * MBb);    // 4MB
    u16* WPOOL   = (u16*)(wsb + 132 * MBb);    // ~1.7MB, 12 weight tensors
    u16* G1_BF   = (u16*)(wsb + 134 * MBb);    // 2MB
    u16* G2_BF   = (u16*)(wsb + 136 * MBb);    // 4MB
    u16* G3_BF   = (u16*)(wsb + 140 * MBb);    // 6MB (lives until DR proj)
    u16* SA      = (u16*)(wsb + 146 * MBb);    // 28MB: XW (GCN) / COL (CNN)
    float* SB    = (float*)(wsb + 174 * MBb);  // 24MB: YG (GCN) / CONVY (CNN)
    u16* C1_BF   = (u16*)(wsb + 198 * MBb);    // 2MB
    u16* C2_BF   = (u16*)(wsb + 200 * MBb);    // 4MB
    u16* CNN_BF  = (u16*)(wsb + 204 * MBb);    // 6MB
    u16* QKV     = (u16*)(wsb + 210 * MBb);    // 6 x 6MB
    u16* SRQ = QKV;               u16* SRK = QKV + 3145728;
    u16* SRV = QKV + 2 * 3145728; u16* DRQ = QKV + 3 * 3145728;
    u16* DRK = QKV + 4 * 3145728; u16* DRV = QKV + 5 * 3145728;
    float* FEAT = (float*)(wsb + 246 * MBb);   // 12MB
    float* POOL = (float*)(wsb + 258 * MBb);   // 12KB
    float* SSR  = (float*)(wsb + 0 * MBb);     // 64MB f32 scores
    float* SDR  = (float*)(wsb + 64 * MBb);    // 64MB f32 scores

    // ----- one-time bf16 conversions --------------------------------------
    cvt(stream, x, X_BF, (long long)B * N * 128);
    cvt(stream, adj, ADJ_BF, (long long)B * NN);
    cvt(stream, degree, DEG_BF, (long long)B * NN);
    const int gwN[3] = {128 * 128, 64 * 256, 128 * 384};
    const int cwN[3] = {128 * 896, 256 * 448, 384 * 896};
    u16* wp = WPOOL;
    u16 *GWB[3], *CWB[3], *PWB[6];
    for (int i = 0; i < 3; ++i) { GWB[i] = wp; cvt(stream, gW[i], wp, gwN[i]); wp += gwN[i]; }
    for (int i = 0; i < 3; ++i) { CWB[i] = wp; cvt(stream, cW[i], wp, cwN[i]); wp += cwN[i]; }
    for (int i = 0; i < 6; ++i) { PWB[i] = wp; cvt(stream, pW[i], wp, C3 * C3); wp += C3 * C3; }

    // ----- normalized adjacency: ADJN = degree @ adj @ degree (bf16) ------
    launch_gemm(stream, false, false, false, true, DEG_BF, N, (long long)NN,
                ADJ_BF, N, (long long)NN, T1_BF, N, (long long)NN, N, N, N, B, nullptr);
    launch_gemm(stream, false, false, false, true, T1_BF, N, (long long)NN,
                DEG_BF, N, (long long)NN, ADJN_BF, N, (long long)NN, N, N, N, B, nullptr);

    // ----- GCN branch ------------------------------------------------------
    const int Fin[3] = {128, 64, 128}, Fout[3] = {128, 256, 384};
    u16* Gin[3]  = {X_BF, G1_BF, G2_BF};
    u16* Gout[3] = {G1_BF, G2_BF, G3_BF};
    for (int l = 0; l < 3; ++l) {
        launch_gemm(stream, false, false, false, true, Gin[l], Fin[l], 0,
                    GWB[l], Fout[l], 0, SA, Fout[l], 0, B * N, Fout[l], Fin[l], 1, nullptr);
        launch_gemm(stream, false, false, false, false, ADJN_BF, N, (long long)NN,
                    SA, Fout[l], (long long)N * Fout[l], SB, Fout[l], (long long)N * Fout[l],
                    N, Fout[l], N, B, nullptr);
        int F = Fout[l] / 2;
        glu_last_k<<<(B * N * F + 255) / 256, 256, 0, stream>>>(SB, gB[l], Gout[l], B * N, F);
    }

    // ----- CNN branch (im2col GEMMs, FL=7 pad=3) --------------------------
    const int Ci[3] = {128, 64, 128}, Co2[3] = {128, 256, 384};
    u16* Cin[3]  = {X_BF, C1_BF, C2_BF};
    u16* Cout[3] = {C1_BF, C2_BF, CNN_BF};
    for (int l = 0; l < 3; ++l) {
        long long tot = (long long)B * Ci[l] * 7 * N;
        im2col_k<<<(unsigned)((tot + 255) / 256), 256, 0, stream>>>(
            Cin[l], SA, B, Ci[l], N, l == 0 ? 1 : 0);
        launch_gemm(stream, false, false, false, false, CWB[l], Ci[l] * 7, 0,
                    SA, N, (long long)(Ci[l] * 7) * N, SB, N, (long long)Co2[l] * N,
                    Co2[l], N, Ci[l] * 7, B, nullptr);
        glu_ch_k<<<(unsigned)(((long long)B * (Co2[l] / 2) * N + 255) / 256), 256, 0, stream>>>(
            SB, cB[l], Cout[l], B, Co2[l], N);
    }

    // ----- 1x1 Q/K/V projections (bf16 out + f32 bias) --------------------
    const long long sP = (long long)C3 * N;
    u16* Pout[6] = {SRQ, SRK, SRV, DRQ, DRK, DRV};
    for (int i = 0; i < 6; ++i) {
        if (i < 3)   // SR*: B = CNN_BF [192,1024] channel-major (normal)
            launch_gemm(stream, false, false, false, true, PWB[i], C3, 0,
                        CNN_BF, N, sP, Pout[i], N, sP, C3, N, C3, B, pB[i]);
        else         // DR*: B = G3 [N,192] -> transposed access
            launch_gemm(stream, false, true, false, true, PWB[i], C3, 0,
                        G3_BF, C3, sP, Pout[i], N, sP, C3, N, C3, B, pB[i]);
    }

    // ----- dual cross-attention -------------------------------------------
    launch_gemm(stream, true, false, false, false, SRQ, N, sP, DRK, N, sP,
                SSR, N, (long long)NN, N, N, C3, B, nullptr);
    launch_gemm(stream, true, false, false, false, DRQ, N, sP, SRK, N, sP,
                SDR, N, (long long)NN, N, N, C3, B, nullptr);
    softmax_pack_k<<<B * N, 256, 0, stream>>>(SSR, N, NORM);   // bf16 in place
    softmax_pack_k<<<B * N, 256, 0, stream>>>(SDR, N, NORM);
    launch_gemm(stream, false, true, false, false, (u16*)SSR, 2 * N, 2 * (long long)NN,
                SRV, N, sP, FEAT, C3, (long long)N * C3, N, C3, N, B, nullptr);
    launch_gemm(stream, false, true, true, false, (u16*)SDR, 2 * N, 2 * (long long)NN,
                DRV, N, sP, FEAT, C3, (long long)N * C3, N, C3, N, B, nullptr);

    // ----- pool + reparametrization head ----------------------------------
    mean_pool_k<<<dim3(C3, B), 128, 0, stream>>>(FEAT, POOL, N, C3);
    head_k<<<(B * C3 + 255) / 256, 256, 0, stream>>>(
        POOL, l1_w, l1_b, l2_w, l2_b, eps, out, B, C3);
}
